// RoutedFFN_62328565399813
// MI455X (gfx1250) — compile-verified, matrix operand order checked by reading, not run on hardware
//
#include <hip/hip_runtime.h>
#include <hip/hip_bf16.h>

typedef __attribute__((ext_vector_type(16))) __bf16 v16bf;
typedef __attribute__((ext_vector_type(4)))  __bf16 v4bf;
typedef __attribute__((ext_vector_type(8)))  float  v8f;
typedef __attribute__((ext_vector_type(4)))  float  v4f;

#define HDIM 1152

// ---------------------------------------------------------------------------
// Kernel 1: per-token router (logits -> softmax probs) + fp32->bf16 convert of x
// One wave (32 lanes) per token.
// ---------------------------------------------------------------------------
__global__ __launch_bounds__(256)
void router_convert_kernel(const float* __restrict__ x,
                           const float* __restrict__ rw,   // [2, HDIM]
                           const float* __restrict__ rb,   // [2]
                           __bf16* __restrict__ xb,        // [Ntok, HDIM]
                           float* __restrict__ probs,      // [Ntok, 2]
                           int ntok)
{
    const int wave = (blockIdx.x * blockDim.x + threadIdx.x) >> 5;
    const int lane = threadIdx.x & 31;
    if (wave >= ntok) return;

    const float* xr = x  + (size_t)wave * HDIM;
    __bf16*      xo = xb + (size_t)wave * HDIM;

    float d0 = 0.0f, d1 = 0.0f;
    for (int i = lane; i < HDIM; i += 32) {
        float v = xr[i];
        xo[i] = (__bf16)v;
        d0 += v * rw[i];
        d1 += v * rw[HDIM + i];
    }
    #pragma unroll
    for (int off = 16; off > 0; off >>= 1) {
        d0 += __shfl_down(d0, off, 32);
        d1 += __shfl_down(d1, off, 32);
    }
    if (lane == 0) {
        d0 += rb[0];
        d1 += rb[1];
        float m  = fmaxf(d0, d1);
        float e0 = __expf(d0 - m);
        float e1 = __expf(d1 - m);
        float inv = 1.0f / (e0 + e1);
        probs[wave * 2 + 0] = e0 * inv;
        probs[wave * 2 + 1] = e1 * inv;
    }
}

// ---------------------------------------------------------------------------
// Kernel 2: fp32 -> bf16 conversion, 4-wide vectorized (n divisible by 4)
// ---------------------------------------------------------------------------
__global__ __launch_bounds__(256)
void f32_to_bf16_kernel(const v4f* __restrict__ in, v4bf* __restrict__ out, int n4)
{
    int i = blockIdx.x * blockDim.x + threadIdx.x;
    const int stride = gridDim.x * blockDim.x;
    for (; i < n4; i += stride)
        out[i] = __builtin_convertvector(in[i], v4bf);
}

// ---------------------------------------------------------------------------
// Kernel 3: tiled WMMA GEMM   D[m,f] = sum_k A[m,k] * W[f,k]   (+ epilogue)
//   A : [M,K] bf16 row-major, staged in DOUBLE-BUFFERED LDS (4x block reuse;
//       prefetch for step i+1 is waited on only after step i's 8 WMMAs,
//       one barrier per K-step)
//   W : [F,K] bf16 row-major, fragments loaded DIRECTLY from global/L2
//       (only 2x intra-block reuse: the 2 M-waves sharing wn load the same rows)
// Tile: 128(M) x 128(F) per block; 8 waves 2(M) x 4(N); wave tile 64 x 32.
// MODE 0: h = GELU_exact(acc + bias)            -> bf16 Hout[M,F]
// MODE 1: out = p0 * (acc + bias)               -> f32 Out[M,F]   (overwrite)
// MODE 2: out += p1 * (acc + bias)              -> f32 Out[M,F]   (accumulate)
// ---------------------------------------------------------------------------
template <int MODE>
__global__ __launch_bounds__(256)
void gemm_wmma_kernel(const __bf16* __restrict__ A,
                      const __bf16* __restrict__ W,
                      const float* __restrict__ bias,
                      const float* __restrict__ probs,
                      __bf16* __restrict__ Hout,
                      float* __restrict__ Out,
                      int M, int K, int F)
{
    constexpr int BM = 128, BK = 32;
    constexpr int LDT = 40;                 // padded LDS row stride (80B, 16B aligned)

    __shared__ __align__(16) __bf16 As[2][BM * LDT];

    const int tid  = threadIdx.x;
    const int bm   = blockIdx.x * BM;       // token base
    const int bn   = blockIdx.y * 128;      // out-feature base

    const int wave = tid >> 5;              // 0..7
    const int lane = tid & 31;
    const int wm   = (wave & 1) * 64;       // wave M offset inside block tile
    const int wn   = (wave >> 1) * 32;      // wave N offset inside block tile
    const int lr   = lane & 15;             // row (A) / column (B) within 16-tile
    const int lk   = (lane >> 4) * 8;       // K sub-offset: 0 or 8

    v8f acc[4][2];
    #pragma unroll
    for (int mi = 0; mi < 4; ++mi)
        #pragma unroll
        for (int ni = 0; ni < 2; ++ni)
            #pragma unroll
            for (int j = 0; j < 8; ++j) acc[mi][ni][j] = 0.0f;

    // A tile copy: 128 rows x 32 bf16 = 512 x 16B chunks, 2 per thread
    const int c0row = tid >> 2, c0ch = tid & 3;
    const int c1    = tid + 256;
    const int c1row = c1 >> 2,  c1ch = c1 & 3;
    const int l0    = c0row * LDT + c0ch * 8;   // LDS elem offsets for the 2 chunks
    const int l1    = c1row * LDT + c1ch * 8;
    const __bf16* ga0 = A + (size_t)(bm + c0row) * K + c0ch * 8;
    const __bf16* ga1 = A + (size_t)(bm + c1row) * K + c1ch * 8;

    // per-lane weight-row pointers (B columns = W rows, contiguous in K)
    const __bf16* wp0 = W + (size_t)(bn + wn + lr)      * K + lk;
    const __bf16* wp1 = W + (size_t)(bn + wn + 16 + lr) * K + lk;

    union Frag { uint4 q[2]; v16bf v; };

    // one K-step of compute from LDS buffer `buf` at column offset k0
    auto compute_step = [&](const __bf16* __restrict__ Abuf, int k0) {
        Frag b[2];
        b[0].q[0] = *(const uint4*)(wp0 + k0);
        b[0].q[1] = *(const uint4*)(wp0 + k0 + 16);
        b[1].q[0] = *(const uint4*)(wp1 + k0);
        b[1].q[1] = *(const uint4*)(wp1 + k0 + 16);

        Frag a[4];
        #pragma unroll
        for (int mi = 0; mi < 4; ++mi) {
            const __bf16* base = Abuf + (wm + mi * 16 + lr) * LDT + lk;
            a[mi].q[0] = *(const uint4*)(base);
            a[mi].q[1] = *(const uint4*)(base + 16);
        }
        #pragma unroll
        for (int mi = 0; mi < 4; ++mi)
            #pragma unroll
            for (int ni = 0; ni < 2; ++ni)
                acc[mi][ni] = __builtin_amdgcn_wmma_f32_16x16x32_bf16(
                    false, a[mi].v, false, b[ni].v,
                    (short)0, acc[mi][ni], false, false);
    };

    const int nsteps = K / BK;

    // prologue: stage tile 0 into buffer 0
    {
        const uint4 a0 = *(const uint4*)(ga0);
        const uint4 a1 = *(const uint4*)(ga1);
        *(uint4*)(&As[0][l0]) = a0;
        *(uint4*)(&As[0][l1]) = a1;
    }
    __syncthreads();

    // steady state: prefetch i+1, compute i, store i+1, one barrier
    for (int i = 0; i < nsteps - 1; ++i) {
        const int k0 = i * BK;
        const uint4 na0 = *(const uint4*)(ga0 + k0 + BK);   // tile i+1
        const uint4 na1 = *(const uint4*)(ga1 + k0 + BK);

        compute_step(As[i & 1], k0);                         // 8 WMMAs hide the loads

        __bf16* dst = As[(i + 1) & 1];
        *(uint4*)(&dst[l0]) = na0;                           // waits loadcnt here
        *(uint4*)(&dst[l1]) = na1;
        __syncthreads();
    }
    // final step (no prefetch, no barrier)
    compute_step(As[(nsteps - 1) & 1], (nsteps - 1) * BK);

    // ---- epilogue --------------------------------------------------------
    // C/D layout: VGPR j, lanes 0-15 -> row j, col lane; lanes 16-31 -> row j+8
    const int rbase = bm + wm + (lane >> 4) * 8;
    #pragma unroll
    for (int mi = 0; mi < 4; ++mi) {
        #pragma unroll
        for (int ni = 0; ni < 2; ++ni) {
            const int col = bn + wn + ni * 16 + lr;
            const float bv = bias[col];
            #pragma unroll
            for (int j = 0; j < 8; ++j) {
                const int row = rbase + mi * 16 + j;
                float v = acc[mi][ni][j] + bv;
                if (MODE == 0) {
                    float g = 0.5f * v * (1.0f + erff(v * 0.70710678118654752f));
                    Hout[(size_t)row * F + col] = (__bf16)g;
                } else {
                    const float p = probs[row * 2 + (MODE == 2 ? 1 : 0)];
                    if (MODE == 2) Out[(size_t)row * F + col] += p * v;
                    else           Out[(size_t)row * F + col]  = p * v;
                }
            }
        }
    }
}

// ---------------------------------------------------------------------------
extern "C" void kernel_launch(void* const* d_in, const int* in_sizes, int n_in,
                              void* d_out, int out_size, void* d_ws, size_t ws_size,
                              hipStream_t stream)
{
    const float* x        = (const float*)d_in[0];
    const float* router_w = (const float*)d_in[1];
    const float* router_b = (const float*)d_in[2];
    const float* light_w1 = (const float*)d_in[3];
    const float* light_b1 = (const float*)d_in[4];
    const float* light_w2 = (const float*)d_in[5];
    const float* light_b2 = (const float*)d_in[6];
    const float* heavy_w1 = (const float*)d_in[7];
    const float* heavy_b1 = (const float*)d_in[8];
    const float* heavy_w2 = (const float*)d_in[9];
    const float* heavy_b2 = (const float*)d_in[10];
    float* out = (float*)d_out;

    const int NT    = in_sizes[0] / HDIM;   // 32768 tokens
    const int LIGHT = in_sizes[4];          // 1152
    const int HEAVY = in_sizes[8];          // 4608

    // ---- workspace carve-up (256B aligned) ----
    char* ws = (char*)d_ws;
    size_t off = 0;
    auto carve = [&](size_t bytes) -> void* {
        off = (off + 255) & ~(size_t)255;
        void* p = ws + off;
        off += bytes;
        return p;
    };
    __bf16* xb    = (__bf16*)carve((size_t)NT * HDIM * 2);
    __bf16* lw1b  = (__bf16*)carve((size_t)LIGHT * HDIM * 2);
    __bf16* lw2b  = (__bf16*)carve((size_t)HDIM * LIGHT * 2);
    __bf16* hw1b  = (__bf16*)carve((size_t)HEAVY * HDIM * 2);
    __bf16* hw2b  = (__bf16*)carve((size_t)HDIM * HEAVY * 2);
    float*  probs = (float*) carve((size_t)NT * 2 * 4);
    __bf16* hl    = (__bf16*)carve((size_t)NT * LIGHT * 2);
    __bf16* hh    = (__bf16*)carve((size_t)NT * HEAVY * 2);

    // 1) router probs + x -> bf16 (one wave per token)
    router_convert_kernel<<<(NT * 32 + 255) / 256, 256, 0, stream>>>(
        x, router_w, router_b, xb, probs, NT);

    // 2) weight conversions (vectorized 4-wide)
    f32_to_bf16_kernel<<<512, 256, 0, stream>>>(
        (const v4f*)light_w1, (v4bf*)lw1b, LIGHT * HDIM / 4);
    f32_to_bf16_kernel<<<512, 256, 0, stream>>>(
        (const v4f*)light_w2, (v4bf*)lw2b, HDIM * LIGHT / 4);
    f32_to_bf16_kernel<<<1024, 256, 0, stream>>>(
        (const v4f*)heavy_w1, (v4bf*)hw1b, HEAVY * HDIM / 4);
    f32_to_bf16_kernel<<<1024, 256, 0, stream>>>(
        (const v4f*)heavy_w2, (v4bf*)hw2b, HDIM * HEAVY / 4);

    // 3) fc1 (+GELU) for both experts
    gemm_wmma_kernel<0><<<dim3(NT / 128, LIGHT / 128), 256, 0, stream>>>(
        xb, lw1b, light_b1, nullptr, hl, nullptr, NT, HDIM, LIGHT);
    gemm_wmma_kernel<0><<<dim3(NT / 128, HEAVY / 128), 256, 0, stream>>>(
        xb, hw1b, heavy_b1, nullptr, hh, nullptr, NT, HDIM, HEAVY);

    // 4) fc2 light: out = p0*(hl@w2 + b2)   (overwrites poisoned d_out)
    gemm_wmma_kernel<1><<<dim3(NT / 128, HDIM / 128), 256, 0, stream>>>(
        hl, lw2b, light_b2, probs, nullptr, out, NT, LIGHT, HDIM);

    // 5) fc2 heavy: out += p1*(hh@w2 + b2)  (same-stream ordering => deterministic)
    gemm_wmma_kernel<2><<<dim3(NT / 128, HDIM / 128), 256, 0, stream>>>(
        hh, hw2b, heavy_b2, probs, nullptr, out, NT, HEAVY, HDIM);
}
